// TGNMemory_58780922413723
// MI455X (gfx1250) — matrix operand order repaired and checked by reading, not compiled, not run on hardware
//
#include <hip/hip_runtime.h>
#include <hip/hip_bf16.h>

typedef __attribute__((ext_vector_type(16))) __bf16 v16bf;
typedef __attribute__((ext_vector_type(8)))  __bf16 v8bf;
typedef __attribute__((ext_vector_type(8)))  float  v8f;

#define MDIM 128      // memory / hidden width
#define MSGW 512      // concat message width
#define G3   384      // 3*M (gate width)
#define LDP  528      // padded LDS msg row, bf16 elems (16B-aligned chunks)

// fast sigmoid: v_exp_f32 + v_rcp_f32 (no IEEE div expansion)
__device__ __forceinline__ float sigmoid_f(float x) {
    return __builtin_amdgcn_rcpf(1.0f + __expf(-x));
}
// fast tanh: (e^{2x}-1)/(e^{2x}+1) via v_exp + v_rcp
__device__ __forceinline__ float tanh_f(float x) {
    float t = __expf(2.0f * x);
    return (t - 1.0f) * __builtin_amdgcn_rcpf(t + 1.0f);
}

// ---------------------------------------------------------------------------
// fp32 -> bf16 weight conversion (once per launch; weights then live in L2)
// ---------------------------------------------------------------------------
__global__ void cvt_bf16_kernel(const float* __restrict__ s, __bf16* __restrict__ d, int n) {
    int i = blockIdx.x * blockDim.x + threadIdx.x;
    if (i < n) d[i] = (__bf16)s[i];
}

// ---------------------------------------------------------------------------
// Fused per-phase kernel: 1 block = 16 edges.
//   stage msg[16x512] (bf16) in LDS, WMMA gi[16x384] & gh[16x384] (f32 acc),
//   GRU epilogue -> h_new[e][128] (f32) in workspace.
// ---------------------------------------------------------------------------
__global__ __launch_bounds__(256)
void tgn_gru_wmma_kernel(const float* __restrict__ mem,      // [N,128] current memory
                         const int*   __restrict__ lu,       // [N] current last_update
                         const int*   __restrict__ aIdx,     // [E] rows being updated
                         const int*   __restrict__ bIdx,     // [E] neighbor rows
                         const int*   __restrict__ tArr,     // [E]
                         const float* __restrict__ rawMsg,   // [E,128]
                         const float* __restrict__ timeW,    // [128]
                         const float* __restrict__ timeB,    // [128]
                         const __bf16* __restrict__ Wih,     // [384,512] bf16
                         const __bf16* __restrict__ Whh,     // [384,128] bf16
                         const float* __restrict__ bIh,      // [384]
                         const float* __restrict__ bHh,      // [384]
                         float* __restrict__ hnew,           // [E,128] out
                         int E) {
    __shared__ __attribute__((aligned(32))) __bf16 msgT[16][LDP]; // bf16 messages
    __shared__ float giL[16][G3];
    __shared__ float ghL[16][G3];
    __shared__ float hF [16][MDIM];                               // fp32 copy of h = mem[a]

    const int tid  = threadIdx.x;
    const int lane = tid & 31;                                   // wave32
    const int wv   = __builtin_amdgcn_readfirstlane(tid >> 5);   // wave id, SGPR-uniform
    const int e0   = blockIdx.x * 16;

    // ---- stage 16 messages: [mem[a] | mem[b] | raw | cos(t*w+b)] as bf16 ----
    const int col = tid * 2;            // 256 threads x 2 cols = 512 cols
    for (int ei = 0; ei < 16; ++ei) {
        int e = e0 + ei;
        if (e >= E) e = E - 1;          // tail clamp (E=100000 is divisible by 16)
        const int ai = aIdx[e];
        float x0, x1;
        if (col < 128) {
            const float* p = mem + (size_t)ai * MDIM + col;
            x0 = p[0]; x1 = p[1];
            hF[ei][col] = x0; hF[ei][col + 1] = x1;
        } else if (col < 256) {
            const int bi = bIdx[e];
            const float* p = mem + (size_t)bi * MDIM + (col - 128);
            x0 = p[0]; x1 = p[1];
        } else if (col < 384) {
            const float* p = rawMsg + (size_t)e * MDIM + (col - 256);
            x0 = p[0]; x1 = p[1];
        } else {
            const float trel = (float)(tArr[e] - lu[ai]);
            const int j = col - 384;
            x0 = __cosf(fmaf(trel, timeW[j],     timeB[j]));
            x1 = __cosf(fmaf(trel, timeW[j + 1], timeB[j + 1]));
        }
        msgT[ei][col]     = (__bf16)x0;
        msgT[ei][col + 1] = (__bf16)x1;
    }
    __syncthreads();

    // ---- WMMA fragment lane mapping (16-bit A 16x32 / B 32x16, wave32) ----
    const int m16   = lane & 15;        // A: edge row / B: output col within tile
    const int khalf = lane >> 4;        // which K-half this lane holds

    // ---- gi = msg[16x512] x Wih^T tile: 3 tiles/wave, K=512 fully unrolled ----
#pragma unroll
    for (int i = 0; i < 3; ++i) {
        const int n0 = (wv + i * 8) * 16;                  // uniform tile column base
        const __bf16* __restrict__ wrow = Wih + (size_t)(n0 + m16) * MSGW + khalf * 16;
        v8f acc = {};
#pragma unroll
        for (int k0 = 0; k0 < MSGW; k0 += 32) {
            union { v16bf v; struct { v8bf lo, hi; } p; } A;
            A.p.lo = *(const v8bf*)&msgT[m16][k0 + khalf * 16];
            A.p.hi = *(const v8bf*)&msgT[m16][k0 + khalf * 16 + 8];
            const v16bf Bv = *(const v16bf*)(wrow + k0);
            acc = __builtin_amdgcn_wmma_f32_16x16x32_bf16(
                false, A.v, false, Bv, (short)0, acc, false, false);
        }
        const int mbase = khalf * 8;
#pragma unroll
        for (int r = 0; r < 8; ++r)
            giL[mbase + r][n0 + m16] = acc[r];
    }

    // ---- gh = h[16x128] x Whh^T tile: 3 tiles/wave, K=128 fully unrolled ----
#pragma unroll
    for (int i = 0; i < 3; ++i) {
        const int n0 = (wv + i * 8) * 16;
        const __bf16* __restrict__ wrow = Whh + (size_t)(n0 + m16) * MDIM + khalf * 16;
        v8f acc = {};
#pragma unroll
        for (int k0 = 0; k0 < MDIM; k0 += 32) {
            union { v16bf v; struct { v8bf lo, hi; } p; } A;
            A.p.lo = *(const v8bf*)&msgT[m16][k0 + khalf * 16];      // h = msg cols 0..127
            A.p.hi = *(const v8bf*)&msgT[m16][k0 + khalf * 16 + 8];
            const v16bf Bv = *(const v16bf*)(wrow + k0);
            acc = __builtin_amdgcn_wmma_f32_16x16x32_bf16(
                false, A.v, false, Bv, (short)0, acc, false, false);
        }
        const int mbase = khalf * 8;
#pragma unroll
        for (int r = 0; r < 8; ++r)
            ghL[mbase + r][n0 + m16] = acc[r];
    }
    __syncthreads();

    // ---- GRU epilogue (fp32): r,z = sigmoid; n = tanh(i_n + r*h_n) ----
    for (int idx = tid; idx < 16 * MDIM; idx += 256) {
        const int ei = idx >> 7;
        const int j  = idx & 127;
        const int e  = e0 + ei;
        if (e >= E) continue;
        const float gr  = giL[ei][j]       + bIh[j]       + ghL[ei][j]       + bHh[j];
        const float gz  = giL[ei][128 + j] + bIh[128 + j] + ghL[ei][128 + j] + bHh[128 + j];
        const float gin = giL[ei][256 + j] + bIh[256 + j];
        const float ghn = ghL[ei][256 + j] + bHh[256 + j];
        const float r = sigmoid_f(gr);
        const float z = sigmoid_f(gz);
        const float n = tanh_f(fmaf(r, ghn, gin));
        const float h = hF[ei][j];
        hnew[(size_t)e * MDIM + j] = fmaf(z, h - n, n);   // (1-z)*n + z*h
    }
}

// ---------------------------------------------------------------------------
// Deterministic scatter: winner[n] = max edge writing node n (matches
// "last duplicate wins"); lu = max-scatter of t.
// ---------------------------------------------------------------------------
__global__ void fill_winner_kernel(int* __restrict__ w, int n) {
    int i = blockIdx.x * blockDim.x + threadIdx.x;
    if (i < n) w[i] = -1;
}

__global__ void vote_kernel(const int* __restrict__ aIdx, const int* __restrict__ tArr,
                            int* __restrict__ winner, int* __restrict__ lu, int E) {
    int e = blockIdx.x * blockDim.x + threadIdx.x;
    if (e < E) {
        atomicMax(&winner[aIdx[e]], e);
        atomicMax(&lu[aIdx[e]], tArr[e]);
    }
}

__global__ void scatter_kernel(const float* __restrict__ hnew, const int* __restrict__ aIdx,
                               const int* __restrict__ winner, float* __restrict__ mem, int E) {
    int idx = blockIdx.x * blockDim.x + threadIdx.x;   // E*32 threads, float4 each
    int e = idx >> 5, q = idx & 31;
    if (e < E) {
        int n = aIdx[e];
        if (winner[n] == e) {
            float4 v = ((const float4*)(hnew + (size_t)e * MDIM))[q];
            ((float4*)(mem + (size_t)n * MDIM))[q] = v;
        }
    }
}

// ---------------------------------------------------------------------------
extern "C" void kernel_launch(void* const* d_in, const int* in_sizes, int n_in,
                              void* d_out, int out_size, void* d_ws, size_t ws_size,
                              hipStream_t stream) {
    const float* memory = (const float*)d_in[0];
    const int*   lastUp = (const int*)  d_in[1];
    const int*   src    = (const int*)  d_in[2];
    const int*   dstI   = (const int*)  d_in[3];
    const int*   tArr   = (const int*)  d_in[4];
    const float* rawMsg = (const float*)d_in[5];
    const float* timeW  = (const float*)d_in[6];
    const float* timeB  = (const float*)d_in[7];
    const float* Wih    = (const float*)d_in[8];
    const float* bIh    = (const float*)d_in[9];
    const float* Whh    = (const float*)d_in[10];
    const float* bHh    = (const float*)d_in[11];

    const int N = in_sizes[1];
    const int E = in_sizes[2];

    float* outMem = (float*)d_out;
    int*   outLu  = (int*)((float*)d_out + (size_t)N * MDIM);

    // workspace carve (all chunks 64B-aligned by construction)
    char* ws = (char*)d_ws;
    float*  hnew   = (float*)ws;  ws += (size_t)E * MDIM * sizeof(float);
    int*    winner = (int*)ws;    ws += (size_t)N * sizeof(int);
    __bf16* WihB   = (__bf16*)ws; ws += (size_t)G3 * MSGW * sizeof(__bf16);
    __bf16* WhhB   = (__bf16*)ws;

    // initialize outputs from inputs (phases update d_out in place)
    hipMemcpyAsync(outMem, memory, (size_t)N * MDIM * sizeof(float),
                   hipMemcpyDeviceToDevice, stream);
    hipMemcpyAsync(outLu, lastUp, (size_t)N * sizeof(int),
                   hipMemcpyDeviceToDevice, stream);

    cvt_bf16_kernel<<<(G3 * MSGW + 255) / 256, 256, 0, stream>>>(Wih, WihB, G3 * MSGW);
    cvt_bf16_kernel<<<(G3 * MDIM + 255) / 256, 256, 0, stream>>>(Whh, WhhB, G3 * MDIM);

    const int nBlocks = (E + 15) / 16;
    for (int p = 0; p < 2; ++p) {
        const int* a = (p == 0) ? src  : dstI;
        const int* b = (p == 0) ? dstI : src;
        tgn_gru_wmma_kernel<<<nBlocks, 256, 0, stream>>>(
            outMem, outLu, a, b, tArr, rawMsg, timeW, timeB,
            WihB, WhhB, bIh, bHh, hnew, E);
        fill_winner_kernel<<<(N + 255) / 256, 256, 0, stream>>>(winner, N);
        vote_kernel<<<(E + 255) / 256, 256, 0, stream>>>(a, tArr, winner, outLu, E);
        scatter_kernel<<<((size_t)E * 32 + 255) / 256, 256, 0, stream>>>(
            hnew, a, winner, outMem, E);
    }
}
